// ProteinGNN_20392504721603
// MI455X (gfx1250) — compile-verified, matrix operand order checked by reading, not run on hardware
//
#include <hip/hip_runtime.h>

#define THREADS 256
#define NEG_SLOPE 0.2f
#define C128 128

typedef __attribute__((ext_vector_type(16))) __bf16 v16bf;
typedef __attribute__((ext_vector_type(8)))  __bf16 v8bf;
typedef __attribute__((ext_vector_type(8)))  float  v8f;

__device__ __forceinline__ float lrelu(float x) { return x > 0.f ? x : NEG_SLOPE * x; }

// float atomic max via signed/unsigned int trick (init value -inf).
__device__ __forceinline__ void atomicMaxFloat(float* addr, float val) {
    if (val >= 0.f) atomicMax((int*)addr, __float_as_int(val));
    else            atomicMin((unsigned int*)addr, __float_as_uint(val));
}

// ---------------------------------------------------------------------------
// Prep: X fp32 [N,F] -> bf16 [Npad,Fp] row-major, zero padded (rows & cols).
// ---------------------------------------------------------------------------
__global__ void pack_x_bf16(const float* __restrict__ X, __bf16* __restrict__ Xb,
                            int N, int F, int Fp, int Npad) {
    long t = (long)blockIdx.x * blockDim.x + threadIdx.x;
    if (t >= (long)Npad * Fp) return;
    int n = (int)(t / Fp), f = (int)(t % Fp);
    float v = (n < N && f < F) ? X[(size_t)n * F + f] : 0.f;
    Xb[t] = (__bf16)v;
}

// ---------------------------------------------------------------------------
// Prep: W fp32 [H,F,C128] -> bf16 swizzled for direct B-fragment loads.
// Wp[(((hd*KT + kt)*8 + nt)*32 + lane)*16 + j], j -> pair i=j>>1,
//   k = kt*32 + (lane>>4)*8 + 2i + (i>=4?8:0) + (j&1), col = nt*16 + (lane&15).
// Zero-pads k >= F (layer 1: F=20 -> Fp=32).
// ---------------------------------------------------------------------------
__global__ void pack_w_bf16(const float* __restrict__ W, __bf16* __restrict__ Wp,
                            int F, int Fp, int Hh) {
    int KT = Fp >> 5;
    long tot = (long)Hh * KT * 4096;        // 8 nt * 32 lane * 16 j
    long t = (long)blockIdx.x * blockDim.x + threadIdx.x;
    if (t >= tot) return;
    int j    = (int)(t & 15);
    int lane = (int)((t >> 4) & 31);
    int nt   = (int)((t >> 9) & 7);
    long rest = t >> 12;
    int kt = (int)(rest % KT);
    int hd = (int)(rest / KT);
    int i = j >> 1;
    int k = (kt << 5) + ((lane >> 4) << 3) + (i << 1) + ((i >= 4) ? 8 : 0) + (j & 1);
    int col = (nt << 4) + (lane & 15);
    float v = (k < F) ? W[((size_t)hd * F + k) * C128 + col] : 0.f;
    Wp[t] = (__bf16)v;
}

// ---------------------------------------------------------------------------
// WMMA GEMM: Hout[n, hd, c] = sum_f Xb[n,f] * W[hd,f,c].  One wave per 16x16
// output tile; K loop in steps of 32. Fragments load as contiguous b128s:
// A: two 16-byte chunks per lane from padded row-major bf16 X.
// B: one 32-byte chunk per lane from the swizzled weight buffer.
// ---------------------------------------------------------------------------
__global__ void gat_gemm_wmma(const __bf16* __restrict__ Xb, const __bf16* __restrict__ Wp,
                              float* __restrict__ Hout, int N, int Fp, int Hh) {
    int lane = threadIdx.x & 31;
    int wave = blockIdx.x * (blockDim.x >> 5) + (threadIdx.x >> 5);
    int mtiles = (N + 15) >> 4;
    int KT = Fp >> 5;
    int total = mtiles * 8 * Hh;
    if (wave >= total) return;           // uniform per wave
    int hd = wave % Hh;
    int t  = wave / Hh;
    int nt = t & 7;
    int mt = t >> 3;
    int r  = lane & 15;
    int hi = lane >> 4;
    int row = (mt << 4) + r;
    int col = (nt << 4) + r;
    const __bf16* xr = Xb + (size_t)row * Fp + (hi << 3);
    const __bf16* wq = Wp + ((size_t)(hd * KT * 8 + nt) * 32 + lane) * 16;

    v8f acc = {};
    for (int kt = 0; kt < KT; ++kt) {
        v8bf alo = *(const v8bf*)(xr);        // k = kbase .. kbase+7
        v8bf ahi = *(const v8bf*)(xr + 16);   // k = kbase+16 .. kbase+23
        v16bf a = __builtin_shufflevector(alo, ahi,
                    0, 1, 2, 3, 4, 5, 6, 7, 8, 9, 10, 11, 12, 13, 14, 15);
        v16bf b = *(const v16bf*)(wq);
        acc = __builtin_amdgcn_wmma_f32_16x16x32_bf16(false, a, false, b,
                                                      (short)0, acc, false, false);
        xr += 32;
        wq += 4096;                           // 8*32*16 bf16 per K-tile
    }
#pragma unroll
    for (int j = 0; j < 8; ++j) {
        int m = (mt << 4) + j + (hi << 3);
        if (m < N) Hout[((size_t)m * Hh + hd) * C128 + col] = acc[j];
    }
}

// Per-node attention scores: s_src[n,hd] = h[n,hd,:].a_src[hd], same for dst.
__global__ void gat_scores(const float* __restrict__ Hf, const float* __restrict__ a_s,
                           const float* __restrict__ a_d, float* __restrict__ ssrc,
                           float* __restrict__ sdst, int N, int Hh) {
    int t = blockIdx.x * blockDim.x + threadIdx.x;
    if (t >= N * Hh) return;
    int hd = t % Hh;
    const float* hp = Hf + (size_t)t * C128;
    const float* as = a_s + hd * C128;
    const float* ad = a_d + hd * C128;
    float ss = 0.f, sd = 0.f;
    for (int c = 0; c < C128; ++c) { float v = hp[c]; ss += v * as[c]; sd += v * ad[c]; }
    ssrc[t] = ss; sdst[t] = sd;
}

__global__ void gat_init(float* __restrict__ emax, float* __restrict__ den,
                         float* __restrict__ acc, float* __restrict__ nsum,
                         int NH, long NHC, int zero_nsum) {
    long i = (long)blockIdx.x * blockDim.x + threadIdx.x;
    if (i < NHC) acc[i] = 0.f;
    if (i < NH) { emax[i] = -__builtin_inff(); den[i] = 0.f; }
    if (zero_nsum && i < C128) nsum[i] = 0.f;
}

// Pass 1: e = leaky_relu(s_src[src] + s_dst[dst]); segment max into emax[dst].
// Edges e >= E are the implicit self loops (src = dst = e - E).
__global__ void gat_edge_max(const int* __restrict__ ei, int E, int N, int Hh,
                             const float* __restrict__ ssrc, const float* __restrict__ sdst,
                             float* __restrict__ ebuf, float* __restrict__ emax) {
    int t = blockIdx.x * blockDim.x + threadIdx.x;
    int tot = (E + N) * Hh;
    if (t >= tot) return;
    int e = t / Hh, hd = t - e * Hh;
    int s = (e < E) ? ei[e]     : (e - E);
    int d = (e < E) ? ei[E + e] : (e - E);
    float v = lrelu(ssrc[s * Hh + hd] + sdst[d * Hh + hd]);
    ebuf[t] = v;
    atomicMaxFloat(&emax[d * Hh + hd], v);
}

// Pass 2: p = exp(e - emax[dst]); segment sum into den[dst].
__global__ void gat_edge_exp(const int* __restrict__ ei, int E, int N, int Hh,
                             float* __restrict__ ebuf, const float* __restrict__ emax,
                             float* __restrict__ den) {
    int t = blockIdx.x * blockDim.x + threadIdx.x;
    int tot = (E + N) * Hh;
    if (t >= tot) return;
    int e = t / Hh, hd = t - e * Hh;
    int d = (e < E) ? ei[E + e] : (e - E);
    float p = expf(ebuf[t] - emax[d * Hh + hd]);
    ebuf[t] = p;
    atomicAdd(&den[d * Hh + hd], p);
}

// Pass 3: acc[dst,hd,:] += alpha * h[src,hd,:]. One thread per 8 channels.
__global__ void gat_edge_scatter(const int* __restrict__ ei, int E, int N, int Hh,
                                 const float* __restrict__ ebuf, const float* __restrict__ den,
                                 const float* __restrict__ Hf, float* __restrict__ acc) {
    long t = (long)blockIdx.x * blockDim.x + threadIdx.x;
    long tot = (long)(E + N) * Hh * (C128 / 8);
    if (t >= tot) return;
    int  c0 = ((int)(t & 15)) << 3;
    long eh = t >> 4;
    int  hd = (int)(eh % Hh);
    long e  = eh / Hh;
    int s = (e < E) ? ei[e]     : (int)(e - E);
    int d = (e < E) ? ei[E + e] : (int)(e - E);
    float alpha = ebuf[eh] / (den[d * Hh + hd] + 1e-16f);
    const float* hs = Hf  + ((size_t)s * Hh + hd) * C128 + c0;
    float*       ap = acc + ((size_t)d * Hh + hd) * C128 + c0;
#pragma unroll
    for (int c = 0; c < 8; ++c) atomicAdd(&ap[c], hs[c] * alpha);
}

// Head-mean + bias + ReLU -> next layer bf16 input (zero-fills padded rows).
__global__ void gat_finalize_mid(const float* __restrict__ acc, const float* __restrict__ bias,
                                 __bf16* __restrict__ xout, int N, int Npad, int Hh) {
    long t = (long)blockIdx.x * blockDim.x + threadIdx.x;
    if (t >= (long)Npad * C128) return;
    int n = (int)(t >> 7), c = (int)(t & 127);
    if (n >= N) { xout[t] = (__bf16)0.f; return; }
    float v = 0.f;
    for (int hd = 0; hd < Hh; ++hd) v += acc[((size_t)n * Hh + hd) * C128 + c];
    v = v / (float)Hh + bias[c];
    xout[t] = (__bf16)(v > 0.f ? v : 0.f);
}

// Last layer (H=1): add bias, accumulate node-sum per channel.
__global__ void gat_finalize_last(const float* __restrict__ acc, const float* __restrict__ bias,
                                  float* __restrict__ nsum, int N) {
    int t = blockIdx.x * blockDim.x + threadIdx.x;
    if (t >= N * C128) return;
    int c = t & 127;
    atomicAdd(&nsum[c], acc[t] + bias[c]);
}

__global__ void gat_writeout(const float* __restrict__ nsum, float* __restrict__ out, int N) {
    int c = threadIdx.x;
    if (c < C128) out[c] = nsum[c] / (float)N;
}

// ---------------------------------------------------------------------------
static inline unsigned nblocks(long n) { return (unsigned)((n + THREADS - 1) / THREADS); }

static void run_gat_layer(const __bf16* Xb, int Fp, int Hh, const __bf16* Wp,
                          const float* as, const float* ad, const float* b,
                          const int* ei, int E, int N, int Npad,
                          float* hbuf, float* acc, float* ssrc, float* sdst,
                          float* emax, float* den, float* ebuf,
                          __bf16* xout /* null => last layer */, float* nsum,
                          hipStream_t stream) {
    long gthread = (long)((N + 15) / 16) * 8 * Hh * 32;   // one wave per tile
    gat_gemm_wmma<<<nblocks(gthread), THREADS, 0, stream>>>(Xb, Wp, hbuf, N, Fp, Hh);
    gat_scores<<<nblocks((long)N * Hh), THREADS, 0, stream>>>(hbuf, as, ad, ssrc, sdst, N, Hh);
    long NHC = (long)N * Hh * C128;
    gat_init<<<nblocks(NHC), THREADS, 0, stream>>>(emax, den, acc, nsum, N * Hh, NHC,
                                                   xout == nullptr ? 1 : 0);
    long EH = (long)(E + N) * Hh;
    gat_edge_max<<<nblocks(EH), THREADS, 0, stream>>>(ei, E, N, Hh, ssrc, sdst, ebuf, emax);
    gat_edge_exp<<<nblocks(EH), THREADS, 0, stream>>>(ei, E, N, Hh, ebuf, emax, den);
    gat_edge_scatter<<<nblocks(EH * (C128 / 8)), THREADS, 0, stream>>>(ei, E, N, Hh,
                                                                       ebuf, den, hbuf, acc);
    if (xout)
        gat_finalize_mid<<<nblocks((long)Npad * C128), THREADS, 0, stream>>>(acc, b, xout,
                                                                             N, Npad, Hh);
    else
        gat_finalize_last<<<nblocks((long)N * C128), THREADS, 0, stream>>>(acc, b, nsum, N);
}

extern "C" void kernel_launch(void* const* d_in, const int* in_sizes, int n_in,
                              void* d_out, int out_size, void* d_ws, size_t ws_size,
                              hipStream_t stream) {
    (void)n_in; (void)out_size; (void)ws_size;
    const float* x   = (const float*)d_in[0];
    const int*   ei  = (const int*)  d_in[1];
    const float* W1  = (const float*)d_in[2];
    const float* as1 = (const float*)d_in[3];
    const float* ad1 = (const float*)d_in[4];
    const float* b1  = (const float*)d_in[5];
    const float* W2  = (const float*)d_in[6];
    const float* as2 = (const float*)d_in[7];
    const float* ad2 = (const float*)d_in[8];
    const float* b2  = (const float*)d_in[9];
    const float* W3  = (const float*)d_in[10];
    const float* as3 = (const float*)d_in[11];
    const float* ad3 = (const float*)d_in[12];
    const float* b3  = (const float*)d_in[13];
    float* out = (float*)d_out;

    const int F_in = 20;
    int N = in_sizes[0] / F_in;        // 20000
    int E = in_sizes[1] / 2;           // 640000
    int Npad = ((N + 15) / 16) * 16;

    // Workspace layout (256B-aligned chunks). ~105 MB total; L2-resident.
    char* wsb = (char*)d_ws;
    size_t off = 0;
    auto alloc = [&](size_t bytes) -> void* {
        off = (off + 255) & ~(size_t)255;
        void* p = wsb + off;
        off += bytes;
        return p;
    };
    float*  hbuf = (float*) alloc((size_t)N * 4 * C128 * 4);
    float*  acc  = (float*) alloc((size_t)N * 4 * C128 * 4);
    float*  ssrc = (float*) alloc((size_t)N * 4 * 4);
    float*  sdst = (float*) alloc((size_t)N * 4 * 4);
    float*  emax = (float*) alloc((size_t)N * 4 * 4);
    float*  den  = (float*) alloc((size_t)N * 4 * 4);
    float*  ebuf = (float*) alloc((size_t)(E + N) * 4 * 4);
    float*  nsum = (float*) alloc(C128 * 4);
    __bf16* xb0  = (__bf16*)alloc((size_t)Npad * 32 * 2);
    __bf16* xbuf = (__bf16*)alloc((size_t)Npad * C128 * 2);
    __bf16* wp1  = (__bf16*)alloc((size_t)4 * 1 * 4096 * 2);
    __bf16* wp2  = (__bf16*)alloc((size_t)4 * 4 * 4096 * 2);
    __bf16* wp3  = (__bf16*)alloc((size_t)1 * 4 * 4096 * 2);

    // One-time packing (re-run every call for determinism; cheap).
    pack_x_bf16<<<nblocks((long)Npad * 32), THREADS, 0, stream>>>(x, xb0, N, F_in, 32, Npad);
    pack_w_bf16<<<nblocks(4L * 1 * 4096), THREADS, 0, stream>>>(W1, wp1, F_in, 32, 4);
    pack_w_bf16<<<nblocks(4L * 4 * 4096), THREADS, 0, stream>>>(W2, wp2, C128, C128, 4);
    pack_w_bf16<<<nblocks(1L * 4 * 4096), THREADS, 0, stream>>>(W3, wp3, C128, C128, 1);

    // Layer 1: [N,20] -> [N,128], 4 heads, ReLU
    run_gat_layer(xb0, 32, 4, wp1, as1, ad1, b1, ei, E, N, Npad,
                  hbuf, acc, ssrc, sdst, emax, den, ebuf, xbuf, nsum, stream);
    // Layer 2: [N,128] -> [N,128], 4 heads, ReLU (xbuf in/out safe: stream-ordered)
    run_gat_layer(xbuf, C128, 4, wp2, as2, ad2, b2, ei, E, N, Npad,
                  hbuf, acc, ssrc, sdst, emax, den, ebuf, xbuf, nsum, stream);
    // Layer 3: [N,128] -> [N,128], 1 head, then node-mean
    run_gat_layer(xbuf, C128, 1, wp3, as3, ad3, b3, ei, E, N, Npad,
                  hbuf, acc, ssrc, sdst, emax, den, ebuf, nullptr, nsum, stream);
    gat_writeout<<<1, 128, 0, stream>>>(nsum, out, N);
}
